// TreeCnnSegmenter_4355096838688
// MI455X (gfx1250) — matrix operand order repaired and checked by reading, not compiled, not run on hardware
//
#include <hip/hip_runtime.h>
#include <hip/hip_bf16.h>

// ---- problem constants (from reference) ----
#define SPREADN   2
#define EMBD      32
#define TAGSN     16
#define NROWS     65536              // N_NODES + 1
#define BATCHN    16
#define NNB       4                  // SPREAD + 2 neighbors per row
#define TILES_PER_BATCH (NROWS / 16) // 4096
#define TOTAL_TILES (BATCHN * TILES_PER_BATCH) // 65536
#define TILE_PAIRS (TOTAL_TILES / 2)           // 32768

typedef __attribute__((ext_vector_type(16))) _Float16 v16h;
typedef __attribute__((ext_vector_type(8)))  _Float16 v8h;
typedef __attribute__((ext_vector_type(8)))  float    v8f;
typedef __attribute__((ext_vector_type(4)))  float    v4f;

// ------------------------------------------------------------------
// Embedding: h0[b,row,:] = f16(emb[x[b,row],:])   (PAD row never occurs)
// 8 halves per thread, fully coalesced (2x b128 load, 1x b128 store).
// ------------------------------------------------------------------
__global__ void __launch_bounds__(256) embed_kernel(
    const int* __restrict__ x, const float* __restrict__ emb,
    _Float16* __restrict__ h0)
{
    size_t j  = ((size_t)blockIdx.x * blockDim.x + threadIdx.x) * 8;
    size_t re = j >> 5;            // (b*NROWS + row)
    int    e0 = (int)(j & 31);
    int    tok = x[re];
    const float* src = emb + (size_t)tok * EMBD + e0;
    _Float16*    dst = h0 + j;
#pragma unroll
    for (int i = 0; i < 8; ++i) dst[i] = (_Float16)src[i];
}

// ------------------------------------------------------------------
// A-fragment loader: 16x32 f16 A matrix for v_wmma_f32_16x16x32_f16.
// Per ISA 7.12.2: lanes 0-15 -> M=lane, halves 0-7 = K 0..7, 8-15 = K 16..23
//                 lanes 16-31 -> M=lane-16, halves 0-7 = K 8..15, 8-15 = K 24..31
// One neighbor row (32 f16 features) == one full K chunk -> two 16B loads.
// ------------------------------------------------------------------
__device__ __forceinline__ v16h load_a_frag(const _Float16* __restrict__ hprev,
                                            size_t batch_off, int nbr_row, int lane)
{
    const _Float16* p = hprev + batch_off + (size_t)nbr_row * EMBD;
    int lo = (lane < 16) ? 0 : 8;
    v8h a0 = *(const v8h*)(p + lo);
    v8h a1 = *(const v8h*)(p + lo + 16);
    v16h a;
#pragma unroll
    for (int i = 0; i < 8; ++i) { a[i] = a0[i]; a[8 + i] = a1[i]; }
    return a;
}

// ------------------------------------------------------------------
// B-fragment builder: 32x16 f16 B matrix (K x N).
// lanes 0-15: half h -> K=h (h<8) / 8+h (h>=8), N=lane ; lanes 16-31: K=8+h (h<8) / 16+h (h>=8), N=lane-16
// Built once per wave from the fp32 weight matrix (stays in L2).
// ------------------------------------------------------------------
template <int OUTDIM>
__device__ __forceinline__ v16h load_b_frag(const float* __restrict__ w,
                                            int lane, int chunk, int ntile)
{
    int n     = (lane & 15) + ntile * 16;
    int kbase = chunk * 32 + ((lane < 16) ? 0 : 8);
    v16h b;
#pragma unroll
    for (int h = 0; h < 16; ++h)
        b[h] = (_Float16)w[(size_t)(kbase + (h < 8 ? h : h + 8)) * OUTDIM + n];
    return b;
}

// ------------------------------------------------------------------
// Hidden layer: out[b,row,:] = leaky_relu( gather(h,idx) @ W(128x32) + bias )
// Each wave owns TWO adjacent 16-row tiles per iteration -> 4 independent
// WMMA accumulator chains (2 tiles x 2 N-tiles): fills the f16 WMMA->WMMA
// hazard slots with independent WMMAs and doubles loads in flight.
// ------------------------------------------------------------------
__global__ void __launch_bounds__(256) layer_kernel(
    const _Float16* __restrict__ hprev, const int* __restrict__ idx,
    const float* __restrict__ w, const float* __restrict__ bias,
    _Float16* __restrict__ hnext, int nwaves_total)
{
    int lane = threadIdx.x & 31;
    int wave = (int)((blockIdx.x * blockDim.x + threadIdx.x) >> 5);

    v16h bfr[2][4];
#pragma unroll
    for (int t = 0; t < 2; ++t)
#pragma unroll
        for (int c = 0; c < 4; ++c)
            bfr[t][c] = load_b_frag<EMBD>(w, lane, c, t);

    float bias0 = bias[lane & 15];
    float bias1 = bias[(lane & 15) + 16];

    for (int pair = wave; pair < TILE_PAIRS; pair += nwaves_total) {
        int    tile0 = pair * 2;                           // tile0,tile0+1 same batch
        int    b     = tile0 >> 12;                        // 4096 tiles per batch
        int    tb0   = (tile0 & (TILES_PER_BATCH - 1)) * 16;
        size_t boff  = (size_t)b * NROWS * EMBD;
        int    row0  = tb0 + (lane & 15);                  // A-matrix M = lane&15
        int    row1  = row0 + 16;

        // prefetch idx row numbered by next pair's tile index (global_prefetch_b8 path)
        __builtin_prefetch(&idx[(size_t)((tile0 + 2 * nwaves_total) & (TOTAL_TILES - 1)) * NNB], 0, 0);

        v8f acc[2][2];
#pragma unroll
        for (int v = 0; v < 8; ++v) {
            acc[0][0][v] = bias0; acc[0][1][v] = bias1;
            acc[1][0][v] = bias0; acc[1][1][v] = bias1;
        }

#pragma unroll
        for (int c = 0; c < 4; ++c) {
            int  nbr0 = idx[(size_t)row0 * NNB + c];
            int  nbr1 = idx[(size_t)row1 * NNB + c];
            v16h a0   = load_a_frag(hprev, boff, nbr0, lane);
            v16h a1   = load_a_frag(hprev, boff, nbr1, lane);
            acc[0][0] = __builtin_amdgcn_wmma_f32_16x16x32_f16(false, a0, false, bfr[0][c],
                                                               (short)0, acc[0][0], false, false);
            acc[1][0] = __builtin_amdgcn_wmma_f32_16x16x32_f16(false, a1, false, bfr[0][c],
                                                               (short)0, acc[1][0], false, false);
            acc[0][1] = __builtin_amdgcn_wmma_f32_16x16x32_f16(false, a0, false, bfr[1][c],
                                                               (short)0, acc[0][1], false, false);
            acc[1][1] = __builtin_amdgcn_wmma_f32_16x16x32_f16(false, a1, false, bfr[1][c],
                                                               (short)0, acc[1][1], false, false);
        }

        // D layout: lane l holds column N=l&15, rows M = v + (l>=16 ? 8 : 0).
        // Per store instruction lanes 0-15 write 32 contiguous bytes of one
        // row (and lanes 16-31 another row) -> coalesced.
        int       n0 = lane & 15;
        int       g8 = (lane < 16) ? 0 : 8;
        _Float16* outp = hnext + boff;
#pragma unroll
        for (int t = 0; t < 2; ++t) {
            int mrow0 = tb0 + t * 16 + g8;
#pragma unroll
            for (int v = 0; v < 8; ++v) {
                float f0 = acc[t][0][v]; f0 = (f0 > 0.f) ? f0 : 0.01f * f0;
                float f1 = acc[t][1][v]; f1 = (f1 > 0.f) ? f1 : 0.01f * f1;
                size_t rb = (size_t)(mrow0 + v) * EMBD;
                outp[rb + n0]      = (_Float16)f0;
                outp[rb + n0 + 16] = (_Float16)f1;
            }
        }
    }
}

// ------------------------------------------------------------------
// Final layer: logits = gather(h,idx) @ W_end(128x16) + b_end, then
// log_softmax over the 16 tags (tags live across 16-lane groups ->
// xor-shuffle reductions), store transposed out[b,tag,row] with two
// 16-byte vector stores per lane per tile.
// ------------------------------------------------------------------
__global__ void __launch_bounds__(256) final_kernel(
    const _Float16* __restrict__ hprev, const int* __restrict__ idx,
    const float* __restrict__ w, const float* __restrict__ bias,
    float* __restrict__ out, int nwaves_total)
{
    int lane = threadIdx.x & 31;
    int wave = (int)((blockIdx.x * blockDim.x + threadIdx.x) >> 5);

    v16h bfr[4];
#pragma unroll
    for (int c = 0; c < 4; ++c) bfr[c] = load_b_frag<TAGSN>(w, lane, c, 0);

    float bias0 = bias[lane & 15];
    int   tag   = lane & 15;
    int   g8    = (lane < 16) ? 0 : 8;

    for (int pair = wave; pair < TILE_PAIRS; pair += nwaves_total) {
        int    tile0 = pair * 2;
        int    b     = tile0 >> 12;
        int    tb0   = (tile0 & (TILES_PER_BATCH - 1)) * 16;
        size_t boff  = (size_t)b * NROWS * EMBD;
        int    row0  = tb0 + (lane & 15);
        int    row1  = row0 + 16;

        v8f acc[2];
#pragma unroll
        for (int v = 0; v < 8; ++v) { acc[0][v] = bias0; acc[1][v] = bias0; }

#pragma unroll
        for (int c = 0; c < 4; ++c) {
            int  nbr0 = idx[(size_t)row0 * NNB + c];
            int  nbr1 = idx[(size_t)row1 * NNB + c];
            v16h a0   = load_a_frag(hprev, boff, nbr0, lane);
            v16h a1   = load_a_frag(hprev, boff, nbr1, lane);
            acc[0] = __builtin_amdgcn_wmma_f32_16x16x32_f16(false, a0, false, bfr[c],
                                                            (short)0, acc[0], false, false);
            acc[1] = __builtin_amdgcn_wmma_f32_16x16x32_f16(false, a1, false, bfr[c],
                                                            (short)0, acc[1], false, false);
        }

#pragma unroll
        for (int t = 0; t < 2; ++t) {
            float yv[8];
#pragma unroll
            for (int v = 0; v < 8; ++v) {
                float val = acc[t][v];
                float mx  = val;
#pragma unroll
                for (int s = 1; s < 16; s <<= 1)
                    mx = fmaxf(mx, __shfl_xor(mx, s, 32));
                float ex  = __expf(val - mx);
                float sum = ex;
#pragma unroll
                for (int s = 1; s < 16; s <<= 1)
                    sum += __shfl_xor(sum, s, 32);
                yv[v] = (val - mx) - __logf(sum);
            }
            // lane l owns 8 consecutive floats of row-range for its tag:
            // out[b, tag, tb0 + 16t + 8g .. +7]  -> two aligned 16B stores
            float* op = out + ((size_t)b * TAGSN + tag) * NROWS
                            + (size_t)(tb0 + t * 16 + g8);
            v4f y0 = { yv[0], yv[1], yv[2], yv[3] };
            v4f y1 = { yv[4], yv[5], yv[6], yv[7] };
            *(v4f*)(op)     = y0;
            *(v4f*)(op + 4) = y1;
        }
    }
}

// ------------------------------------------------------------------
// Launch: embed -> layer1 -> layer2 -> final  (f16 ping-pong in d_ws)
// ------------------------------------------------------------------
extern "C" void kernel_launch(void* const* d_in, const int* in_sizes, int n_in,
                              void* d_out, int out_size, void* d_ws, size_t ws_size,
                              hipStream_t stream)
{
    (void)in_sizes; (void)n_in; (void)out_size; (void)ws_size;

    const int*   x     = (const int*)d_in[0];
    const int*   idx   = (const int*)d_in[1];
    const float* emb   = (const float*)d_in[2];
    const float* w1    = (const float*)d_in[3];
    const float* b1    = (const float*)d_in[4];
    const float* w2    = (const float*)d_in[5];
    const float* b2    = (const float*)d_in[6];
    const float* w_end = (const float*)d_in[7];
    const float* b_end = (const float*)d_in[8];
    float*       out   = (float*)d_out;

    _Float16* hA = (_Float16*)d_ws;                                // 64 MiB
    _Float16* hB = hA + (size_t)BATCHN * NROWS * EMBD;             // 64 MiB

    // embed: 16*65536*32 halves / 8 per thread / 256 per block = 16384 blocks
    embed_kernel<<<16384, 256, 0, stream>>>(x, emb, hA);

    const int blocks = 512;           // 4096 waves; 32768 tile-pairs -> 8 pairs/wave
    const int nwaves = blocks * (256 / 32);

    layer_kernel<<<blocks, 256, 0, stream>>>(hA, idx, w1, b1, hB, nwaves);
    layer_kernel<<<blocks, 256, 0, stream>>>(hB, idx, w2, b2, hA, nwaves);
    final_kernel<<<blocks, 256, 0, stream>>>(hA, idx, w_end, b_end, out, nwaves);
}
